// LoRAWrapperLayer_1108101562874
// MI455X (gfx1250) — matrix operand order repaired and checked by reading, not compiled, not run on hardware
//
#include <hip/hip_runtime.h>
#include <hip/hip_bf16.h>

// ---------------------------------------------------------------------------
// LoRA wrapper layer, folded:  out = x @ (W + s*B@A)^T + b
// Step 0: Xb(bf16)  = bf16(x)                                (ws, 128 MB)
// Step 1: Weff(bf16)= bf16(W + 2 * lora_B @ lora_A)          (ws,  32 MB)
// Step 2: out = Xb @ Weff^T + b   via v_wmma_f32_16x16x32_bf16,
//         double-buffered with global_load_async_to_lds_b128 (ASYNCcnt).
// ---------------------------------------------------------------------------

typedef __bf16 bf16;
typedef __attribute__((ext_vector_type(16))) __bf16 v16bf;
typedef __attribute__((ext_vector_type(8)))  float  v8f;

#define D_IN   4096
#define D_OUT  4096
#define M_TOT  16384   // 4 * 4096
#define RANK   16
#define SCALE  2.0f

// ---------------------------------------------------------------------------
// Kernel 0: cast x (fp32) -> Xb (bf16). 8 elements per thread.
// ---------------------------------------------------------------------------
__global__ __launch_bounds__(256) void cast_x_kernel(
    const float* __restrict__ X, bf16* __restrict__ Xb)
{
    size_t idx = ((size_t)blockIdx.x * 256 + threadIdx.x) * 8;
    float4 a = *reinterpret_cast<const float4*>(X + idx);
    float4 b = *reinterpret_cast<const float4*>(X + idx + 4);
    struct alignas(16) bf16x8 { bf16 v[8]; };
    bf16x8 p;
    p.v[0] = (bf16)a.x; p.v[1] = (bf16)a.y; p.v[2] = (bf16)a.z; p.v[3] = (bf16)a.w;
    p.v[4] = (bf16)b.x; p.v[5] = (bf16)b.y; p.v[6] = (bf16)b.z; p.v[7] = (bf16)b.w;
    *reinterpret_cast<bf16x8*>(Xb + idx) = p;
}

// ---------------------------------------------------------------------------
// Kernel 1: fold the rank-16 LoRA update into the weights, store as bf16.
// ---------------------------------------------------------------------------
__global__ __launch_bounds__(256) void fold_weff_kernel(
    const float* __restrict__ W,
    const float* __restrict__ lora_A,   // [RANK, D_IN]
    const float* __restrict__ lora_B,   // [D_OUT, RANK]
    bf16* __restrict__ Weff)            // [D_OUT, D_IN]
{
    int idx = (blockIdx.x * 256 + threadIdx.x) * 4;
    int o = idx >> 12;
    int k = idx & 4095;

    float4 w = *reinterpret_cast<const float4*>(W + (size_t)o * D_IN + k);
    float a0 = w.x, a1 = w.y, a2 = w.z, a3 = w.w;

    const float* Brow = lora_B + (size_t)o * RANK;
#pragma unroll
    for (int r = 0; r < RANK; ++r) {
        float s = SCALE * Brow[r];
        float4 av = *reinterpret_cast<const float4*>(lora_A + (size_t)r * D_IN + k);
        a0 = fmaf(s, av.x, a0);
        a1 = fmaf(s, av.y, a1);
        a2 = fmaf(s, av.z, a2);
        a3 = fmaf(s, av.w, a3);
    }

    struct alignas(8) bf16x4 { bf16 x, y, z, w; };
    bf16x4 outv;
    outv.x = (bf16)a0; outv.y = (bf16)a1; outv.z = (bf16)a2; outv.w = (bf16)a3;
    *reinterpret_cast<bf16x4*>(Weff + (size_t)o * D_IN + k) = outv;
}

// ---------------------------------------------------------------------------
// Kernel 2: bf16 WMMA GEMM, async-to-LDS double-buffered pipeline.
//   Block tile: 128 (M) x 128 (N), K-step 64, 8 waves: 2(M) x 4(N),
//   each wave: 64x32 = 4x2 fragments = 16 WMMA per K-step.
// ---------------------------------------------------------------------------
#define BLK_M 128
#define BLK_N 128
#define BLK_K 64
#define LDS_PITCH  (BLK_K + 8)          // 72 bf16 = 144 B: 16B-aligned, conflict-free
#define TILE_ELEMS (BLK_M * LDS_PITCH)  // per buffer

union Frag { uint4 u[2]; v16bf v; };

// async 16B global -> LDS copy (ASYNCcnt-tracked), inline asm per CDNA5 ISA.
__device__ __forceinline__ void async_copy16(unsigned int lds_byte, const bf16* gptr) {
    asm volatile("global_load_async_to_lds_b128 %0, %1, off"
                 :: "v"(lds_byte), "v"((unsigned long long)(uintptr_t)gptr)
                 : "memory");
}
#define WAIT_ASYNC(n) asm volatile("s_wait_asynccnt %0" :: "n"(n) : "memory")

__global__ __launch_bounds__(256) void lora_gemm_kernel(
    const bf16*  __restrict__ Xb,     // [M_TOT, D_IN] bf16
    const bf16*  __restrict__ Wb,     // [D_OUT, D_IN] bf16 (W_eff)
    const float* __restrict__ bias,   // [D_OUT]
    float* __restrict__ Out)          // [M_TOT, D_OUT]
{
    __shared__ bf16 As[2 * TILE_ELEMS];   // 2 x 18 KB
    __shared__ bf16 Bs[2 * TILE_ELEMS];   // 2 x 18 KB

    const int tid    = threadIdx.x;
    const int lane   = tid & 31;
    const int wave   = tid >> 5;
    const int wave_m = wave >> 2;         // 0..1
    const int wave_n = wave & 3;          // 0..3

    const int mBase = blockIdx.y * BLK_M;
    const int nBase = blockIdx.x * BLK_N;

    const int lhalf  = lane >> 4;
    const int lmod   = lane & 15;
    const int a_koff = lhalf * 8;         // A: k halves split at lane 16
    const int b_koff = lhalf * 8;         // B: k halves split at lane 16

    // LDS byte offsets for async stores (flat addr[31:0] == wave LDS offset).
    const unsigned int asBase = (unsigned int)(uintptr_t)&As[0];
    const unsigned int bsBase = (unsigned int)(uintptr_t)&Bs[0];

    // Per-thread staging coordinates: 4 x 16B chunks per tile per matrix.
    int st_row[4], st_c8[4];
#pragma unroll
    for (int it = 0; it < 4; ++it) {
        int g = tid + it * 256;           // 0..1023
        st_row[it] = g >> 3;              // 8 x 16B groups per 64-elem row
        st_c8[it]  = (g & 7) << 3;        // bf16 element offset in row
    }

    v8f acc[4][2];
#pragma unroll
    for (int i = 0; i < 4; ++i)
#pragma unroll
        for (int j = 0; j < 2; ++j) { v8f z = {}; acc[i][j] = z; }

    constexpr int NT = D_IN / BLK_K;      // 64 K-tiles

    // Issue tile 0 into buffer 0 (8 async instructions per wave).
#pragma unroll
    for (int it = 0; it < 4; ++it) {
        async_copy16(asBase + (unsigned)(st_row[it] * LDS_PITCH + st_c8[it]) * 2,
                     Xb + (size_t)(mBase + st_row[it]) * D_IN + st_c8[it]);
        async_copy16(bsBase + (unsigned)(st_row[it] * LDS_PITCH + st_c8[it]) * 2,
                     Wb + (size_t)(nBase + st_row[it]) * D_IN + st_c8[it]);
    }

#pragma unroll 2
    for (int t = 0; t < NT; ++t) {
        const int buf = t & 1;

        if (t + 1 < NT) {
            // Prefetch tile t+1 into the other buffer (its previous readers
            // finished at the trailing barrier of iteration t-1).
            const int kt = (t + 1) * BLK_K;
            const unsigned bufOff = (unsigned)((buf ^ 1) * TILE_ELEMS) * 2;
#pragma unroll
            for (int it = 0; it < 4; ++it) {
                async_copy16(asBase + bufOff + (unsigned)(st_row[it] * LDS_PITCH + st_c8[it]) * 2,
                             Xb + (size_t)(mBase + st_row[it]) * D_IN + kt + st_c8[it]);
                async_copy16(bsBase + bufOff + (unsigned)(st_row[it] * LDS_PITCH + st_c8[it]) * 2,
                             Wb + (size_t)(nBase + st_row[it]) * D_IN + kt + st_c8[it]);
            }
            WAIT_ASYNC(8);   // tile t complete (in-order), t+1 may be in flight
        } else {
            WAIT_ASYNC(0);   // last tile: drain
        }
        __syncthreads();

        const bf16* aBuf = &As[buf * TILE_ELEMS];
        const bf16* bBuf = &Bs[buf * TILE_ELEMS];

#pragma unroll
        for (int kk = 0; kk < 2; ++kk) {
            Frag a[4];
#pragma unroll
            for (int i = 0; i < 4; ++i) {
                int row = wave_m * 64 + i * 16 + lmod;
                const bf16* p = &aBuf[row * LDS_PITCH + kk * 32 + a_koff];
                a[i].u[0] = *reinterpret_cast<const uint4*>(p);       // k .. k+7
                a[i].u[1] = *reinterpret_cast<const uint4*>(p + 16);  // k+16 .. k+23
            }
            Frag bfr[2];
#pragma unroll
            for (int j = 0; j < 2; ++j) {
                int col = wave_n * 32 + j * 16 + lmod;
                const bf16* p = &bBuf[col * LDS_PITCH + kk * 32 + b_koff];
                bfr[j].u[0] = *reinterpret_cast<const uint4*>(p);     // k .. k+7
                bfr[j].u[1] = *reinterpret_cast<const uint4*>(p + 16); // k+16 .. k+23
            }
#pragma unroll
            for (int i = 0; i < 4; ++i)
#pragma unroll
                for (int j = 0; j < 2; ++j)
                    acc[i][j] = __builtin_amdgcn_wmma_f32_16x16x32_bf16(
                        false, a[i].v, false, bfr[j].v,
                        (short)0, acc[i][j], false, false);
        }
        __syncthreads();
    }

    // ---- Epilogue: add bias, store fp32 ----
    float bv[2];
#pragma unroll
    for (int j = 0; j < 2; ++j)
        bv[j] = bias[nBase + wave_n * 32 + j * 16 + lmod];

#pragma unroll
    for (int i = 0; i < 4; ++i)
#pragma unroll
        for (int j = 0; j < 2; ++j) {
            int colg = nBase + wave_n * 32 + j * 16 + lmod;
#pragma unroll
            for (int v = 0; v < 8; ++v) {
                int rowg = mBase + wave_m * 64 + i * 16 + v + (lhalf << 3);
                Out[(size_t)rowg * D_OUT + colg] = acc[i][j][v] + bv[j];
            }
        }
}

// ---------------------------------------------------------------------------
// Launch
// ---------------------------------------------------------------------------
extern "C" void kernel_launch(void* const* d_in, const int* in_sizes, int n_in,
                              void* d_out, int out_size, void* d_ws, size_t ws_size,
                              hipStream_t stream) {
    const float* x      = (const float*)d_in[0];
    const float* W      = (const float*)d_in[1];
    const float* b      = (const float*)d_in[2];
    const float* lora_A = (const float*)d_in[3];
    const float* lora_B = (const float*)d_in[4];
    float* out = (float*)d_out;

    bf16* Weff = (bf16*)d_ws;                                  // 32 MB
    bf16* Xb   = (bf16*)((char*)d_ws + (size_t)D_OUT * D_IN * 2); // 128 MB

    {   // Step 0: cast x -> bf16
        size_t total = (size_t)M_TOT * D_IN / 8;
        cast_x_kernel<<<dim3(total / 256), dim3(256), 0, stream>>>(x, Xb);
    }
    {   // Step 1: fold LoRA into weights (bf16)
        int total = D_OUT * D_IN / 4;
        fold_weff_kernel<<<dim3(total / 256), dim3(256), 0, stream>>>(W, lora_A, lora_B, Weff);
    }
    {   // Step 2: WMMA GEMM + bias
        dim3 grid(D_OUT / BLK_N, M_TOT / BLK_M);               // (32, 128)
        lora_gemm_kernel<<<grid, dim3(256), 0, stream>>>(Xb, Weff, b, out);
    }
}
